// DAN_MoVe_85985245266011
// MI455X (gfx1250) — matrix operand order repaired and checked by reading, not compiled, not run on hardware
//
#include <hip/hip_runtime.h>

// ---------------------------------------------------------------------------
// DAN/MoVe attention block for MI455X (gfx1250, wave32, WMMA).
// Shapes: B=2, F=5, SF=15, C=1024, KD=128, VD=512, H=W=32, hw=1024.
// All GEMM-shaped work (4 convs, 1x1 conv, both attention stages) runs on
// v_wmma_f32_16x16x32_f16; conv is register-blocked 32px x 64cout per wave.
// ---------------------------------------------------------------------------

typedef __attribute__((ext_vector_type(16))) _Float16 v16h;
typedef __attribute__((ext_vector_type(8)))  _Float16 v8h;
typedef __attribute__((ext_vector_type(8)))  float    v8f;

#define HW    1024
#define KDIM  128
#define VDIM  512
#define BB    2
#define FF    5
#define SFN   15

// ---- WMMA fragment loaders (ISA 7.12.2 layouts, wave32) -------------------
// A matrix 16x32 f16: lane L -> row M = L%16, half h = L/16,
//   VGPR0..3 hold K = 8h..8h+7, VGPR4..7 hold K = 16+8h..16+8h+7.
__device__ __forceinline__ v16h ld_fragA(const _Float16* __restrict__ row, int c0h) {
  union { v16h v; v8h h[2]; } u;
  u.h[0] = *(const v8h*)(row + c0h);        // K = c0 + 8h .. +7
  u.h[1] = *(const v8h*)(row + c0h + 16);   // K = c0 + 16 + 8h .. +7
  return u.v;
}
// B matrix 32x16 f16, loaded from B^T stored row-major [N rows][K contig]:
// lane L -> col N = L%16, half h = L/16 holds K = k0+8h..+7, k0+16+8h..+7.
__device__ __forceinline__ v16h ld_fragB(const _Float16* __restrict__ rowT, int k0, int half) {
  union { v16h v; v8h h[2]; } u;
  u.h[0] = *(const v8h*)(rowT + k0 + 8 * half);
  u.h[1] = *(const v8h*)(rowT + k0 + 8 * half + 16);
  return u.v;
}

__device__ __forceinline__ v8f wmma16(v16h a, v16h b, v8f c) {
  return __builtin_amdgcn_wmma_f32_16x16x32_f16(false, a, false, b, (short)0, c,
                                                false, false);
}

// ---------------------------------------------------------------------------
// Layout conversion: NCHW f32 -> NHWC f16  (C contiguous = GEMM K dim)
// ---------------------------------------------------------------------------
__global__ void k_nchw2nhwc(const float* __restrict__ in, _Float16* __restrict__ out,
                            int N, int C) {
  long total = (long)N * C * HW;
  long idx = (long)blockIdx.x * blockDim.x + threadIdx.x;
  if (idx >= total) return;
  int p = (int)(idx % HW);
  long t = idx / HW;
  int c = (int)(t % C);
  int n = (int)(t / C);
  out[((long)n * HW + p) * C + c] = (_Float16)in[idx];
}

// Weights [Cout][C][kh][kw] f32 -> [tap][Cout][C] f16 (C contiguous)
__global__ void k_wcvt(const float* __restrict__ w, _Float16* __restrict__ out,
                       int Cout, int C, int taps) {
  long total = (long)Cout * C * taps;
  long idx = (long)blockIdx.x * blockDim.x + threadIdx.x;
  if (idx >= total) return;
  int t  = (int)(idx % taps);
  long r = idx / taps;
  int c  = (int)(r % C);
  int co = (int)(r / C);
  out[((long)t * Cout + co) * C + c] = (_Float16)w[idx];
}

// ---------------------------------------------------------------------------
// Implicit-GEMM conv (3x3 pad 1, or 1x1) via WMMA, register-blocked:
// one wave -> one image row (32 pixels, 2 M-tiles) x 64 Couts (4 N-tiles),
// 8 accumulators; per 32-channel K-step: 2 A frags + 4 B frags -> 8 WMMAs.
// Output address: (i/img_div)*os_hi + (i%img_div)*os_lo + co*os_ch + p*os_pix
// ---------------------------------------------------------------------------
__global__ __launch_bounds__(32)
void k_conv_wmma(const _Float16* __restrict__ in,   // [n][hw][C] f16
                 const _Float16* __restrict__ wt,   // [taps][Cout][C] f16
                 const float* __restrict__ bias,    // [Cout]
                 _Float16* __restrict__ out16, float* __restrict__ out32,
                 int C, int Cout, int ntaps,
                 int img_base, int img_step, int img_div,
                 long os_hi, long os_lo, long os_ch, long os_pix) {
  const int lane = threadIdx.x;
  const int lm = lane & 15, half = lane >> 4;
  const int i = blockIdx.y;
  const int n = img_base + img_step * i;
  const int y0 = blockIdx.x & 31;        // image row (32 rows)
  const int ng = blockIdx.x >> 5;        // 64-Cout group
  const _Float16* img = in + (long)n * HW * C;

  v8f acc0[4], acc1[4];                  // [N-subtile] for M-tile 0 / 1
  v8f zro = {};
#pragma unroll
  for (int s = 0; s < 4; ++s) { acc0[s] = zro; acc1[s] = zro; }
  v16h az = {};

  for (int tap = 0; tap < ntaps; ++tap) {
    int dy = 0, dx = 0;
    if (ntaps == 9) { dy = tap / 3 - 1; dx = tap % 3 - 1; }
    const int py = y0 + dy;
    const bool rowok = (py >= 0) && (py < 32);
    const int px0 = lm + dx;             // M-tile 0: pixels x = 0..15 (+dx)
    const int px1 = 16 + lm + dx;        // M-tile 1: pixels x = 16..31 (+dx)
    const bool v0 = rowok && (px0 >= 0) && (px0 < 32);
    const bool v1 = rowok && (px1 >= 0) && (px1 < 32);
    const _Float16* arow0 = img + (long)(py * 32 + px0) * C;
    const _Float16* arow1 = img + (long)(py * 32 + px1) * C;
    const _Float16* w0 = wt + ((long)tap * Cout + ng * 64 + lm) * C;
    const _Float16* w1 = w0 + 16L * C;
    const _Float16* w2 = w0 + 32L * C;
    const _Float16* w3 = w0 + 48L * C;
    for (int c0 = 0; c0 < C; c0 += 32) {
      v16h a0, a1;
      if (v0) a0 = ld_fragA(arow0, c0 + 8 * half); else a0 = az;
      if (v1) a1 = ld_fragA(arow1, c0 + 8 * half); else a1 = az;
      v16h b0 = ld_fragB(w0, c0, half);
      v16h b1 = ld_fragB(w1, c0, half);
      v16h b2 = ld_fragB(w2, c0, half);
      v16h b3 = ld_fragB(w3, c0, half);
      acc0[0] = wmma16(a0, b0, acc0[0]);
      acc1[0] = wmma16(a1, b0, acc1[0]);
      acc0[1] = wmma16(a0, b1, acc0[1]);
      acc1[1] = wmma16(a1, b1, acc1[1]);
      acc0[2] = wmma16(a0, b2, acc0[2]);
      acc1[2] = wmma16(a1, b2, acc1[2]);
      acc0[3] = wmma16(a0, b3, acc0[3]);
      acc1[3] = wmma16(a1, b3, acc1[3]);
    }
  }

  const long ibase = (long)(i / img_div) * os_hi + (long)(i % img_div) * os_lo;
#pragma unroll
  for (int s = 0; s < 4; ++s) {
    const int co = ng * 64 + s * 16 + lm;
    const float bv = bias[co];
    const long base = ibase + (long)co * os_ch;
#pragma unroll
    for (int r = 0; r < 8; ++r) {        // D row m = r + 8*half, col = lm
      const int m = r + 8 * half;
      const int pA = y0 * 32 + m;        // M-tile 0 pixel
      const int pB = y0 * 32 + 16 + m;   // M-tile 1 pixel
      const float vA = acc0[s][r] + bv;
      const float vB = acc1[s][r] + bv;
      if (out32) {
        out32[base + (long)pA * os_pix] = vA;
        out32[base + (long)pB * os_pix] = vB;
      } else {
        out16[base + (long)pA * os_pix] = (_Float16)vA;
        out16[base + (long)pB * os_pix] = (_Float16)vB;
      }
    }
  }
}

// ---------------------------------------------------------------------------
// Flash attention (both stages). One wave: 16 queries x 128 V-channels.
// S = Q K^T (WMMA), online softmax over keys (shfl_xor row reductions in the
// 16-lane D-layout half-groups), P re-fragmented through LDS, O += P V (WMMA).
// stage1: out16 = new_V [B][VD][hw] f16 (v-major). stage2: out32 = d_out f32.
// ---------------------------------------------------------------------------
__global__ __launch_bounds__(32)
void k_flash_wmma(const _Float16* __restrict__ Q, long q_bs,   // rows [.][KD]
                  const _Float16* __restrict__ K, long k_bs,   // rows [.][KD]
                  const _Float16* __restrict__ V, long v_bs, long v_rs, // [VD][nk]
                  int nk,
                  _Float16* __restrict__ out16,
                  float* __restrict__ out32) {
  __shared__ _Float16 plds[16 * 32];             // P tile: 16 q x 32 k (f16)
  const int lane = threadIdx.x;
  const int lm = lane & 15, half = lane >> 4;
  const int qt = blockIdx.x;
  const int vg = blockIdx.y;                     // 4 groups of 128 V channels
  const int b  = blockIdx.z;
  const float scale = 0.088388347648318447f;     // 1/sqrt(128)

  const _Float16* qrow = Q + (long)b * q_bs + (long)(qt * 16 + lm) * KDIM;
  v16h qf[4];
#pragma unroll
  for (int j = 0; j < 4; ++j) qf[j] = ld_fragA(qrow, 32 * j + 8 * half);

  v8f acc[8];
  v8f zro = {};
#pragma unroll
  for (int vc = 0; vc < 8; ++vc) acc[vc] = zro;
  float mrow[8], lrow[8];
#pragma unroll
  for (int r = 0; r < 8; ++r) { mrow[r] = -1e30f; lrow[r] = 0.0f; }

  const _Float16* Kb = K + (long)b * k_bs;
  const _Float16* Vb = V + (long)b * v_bs + (long)(vg * 128) * v_rs;

  for (int kt = 0; kt < nk; kt += 32) {
    // ---- S tiles: S0 = keys kt..kt+15, S1 = keys kt+16..kt+31 (cols) ----
    const _Float16* k0 = Kb + (long)(kt + lm) * KDIM;        // B^T row = key
    const _Float16* k1 = Kb + (long)(kt + 16 + lm) * KDIM;
    v8f s0 = {}, s1 = {};
#pragma unroll
    for (int j = 0; j < 4; ++j) {
      s0 = wmma16(qf[j], ld_fragB(k0, 32 * j, half), s0);
      s1 = wmma16(qf[j], ld_fragB(k1, 32 * j, half), s1);
    }
    // ---- online softmax (row m = r + 8*half spans 16 lanes of one half) --
    float p0[8], p1[8], cf[8];
#pragma unroll
    for (int r = 0; r < 8; ++r) {
      float a = s0[r] * scale, c = s1[r] * scale;
      float t = fmaxf(a, c);
#pragma unroll
      for (int o = 1; o < 16; o <<= 1) t = fmaxf(t, __shfl_xor(t, o));
      float mn = fmaxf(mrow[r], t);
      cf[r] = __expf(mrow[r] - mn);
      mrow[r] = mn;
      p0[r] = __expf(a - mn);
      p1[r] = __expf(c - mn);
      float rs = p0[r] + p1[r];
#pragma unroll
      for (int o = 1; o < 16; o <<= 1) rs += __shfl_xor(rs, o);
      lrow[r] = lrow[r] * cf[r] + rs;
    }
#pragma unroll
    for (int vc = 0; vc < 8; ++vc)
#pragma unroll
      for (int r = 0; r < 8; ++r) acc[vc][r] *= cf[r];
    // ---- re-fragment P: D-layout f32 -> LDS [16q][32k] -> A-layout f16 ---
#pragma unroll
    for (int r = 0; r < 8; ++r) {
      const int m = r + 8 * half;
      plds[m * 32 + lm]      = (_Float16)p0[r];
      plds[m * 32 + 16 + lm] = (_Float16)p1[r];
    }
    v16h pf = ld_fragA((const _Float16*)plds + lm * 32, 8 * half);
    // ---- O += P * V  (V^T rows are v-major, keys contiguous) -------------
#pragma unroll
    for (int vc = 0; vc < 8; ++vc) {
      const _Float16* vrow = Vb + (long)(vc * 16 + lm) * v_rs + kt;
      acc[vc] = wmma16(pf, ld_fragB(vrow, 0, half), acc[vc]);
    }
  }

#pragma unroll
  for (int r = 0; r < 8; ++r) lrow[r] = 1.0f / lrow[r];
  if (out16) {                                    // stage 1 -> new_V f16
    _Float16* ob = out16 + (long)b * VDIM * HW;
#pragma unroll
    for (int vc = 0; vc < 8; ++vc)
#pragma unroll
      for (int r = 0; r < 8; ++r) {
        const int q = qt * 16 + r + 8 * half;
        const int v = vg * 128 + vc * 16 + lm;
        ob[(long)v * HW + q] = (_Float16)(acc[vc][r] * lrow[r]);
      }
  } else {                                        // stage 2 -> d_out f32
#pragma unroll
    for (int vc = 0; vc < 8; ++vc)
#pragma unroll
      for (int r = 0; r < 8; ++r) {
        const int qg = qt * 16 + r + 8 * half;    // in [0, F*hw)
        const int f = qg >> 10, p = qg & 1023;
        const int v = vg * 128 + vc * 16 + lm;
        out32[((long)(b * FF + f)) * (2L * VDIM * HW) + (long)v * HW + p] =
            acc[vc][r] * lrow[r];
      }
  }
}

// ---------------------------------------------------------------------------
extern "C" void kernel_launch(void* const* d_in, const int* in_sizes, int n_in,
                              void* d_out, int out_size, void* d_ws, size_t ws_size,
                              hipStream_t stream) {
  const float* query   = (const float*)d_in[0];   // [10,1024,32,32]
  const float* support = (const float*)d_in[1];   // [30,1024,32,32]
  const float* w_sk = (const float*)d_in[2];
  const float* b_sk = (const float*)d_in[3];
  const float* w_sv = (const float*)d_in[4];
  const float* b_sv = (const float*)d_in[5];
  const float* w_qq = (const float*)d_in[6];
  const float* b_qq = (const float*)d_in[7];
  const float* w_qk = (const float*)d_in[8];
  const float* b_qk = (const float*)d_in[9];
  const float* w_cq = (const float*)d_in[10];
  const float* b_cq = (const float*)d_in[11];
  float* out = (float*)d_out;                     // [10, 1024, 32, 32]

  // workspace carve-out (~147 MB, 256B aligned slices)
  char* ws = (char*)d_ws;
  size_t off = 0;
  auto take = [&](size_t elems) -> _Float16* {
    _Float16* p = (_Float16*)(ws + off);
    off += (elems * 2 + 255) & ~(size_t)255;
    return p;
  };
  _Float16* q16  = take(10ull * 1024 * HW);        // query NHWC f16
  _Float16* s16  = take(30ull * 1024 * HW);        // support NHWC f16
  _Float16* wsk  = take(9ull * 128 * 1024);
  _Float16* wsv  = take(9ull * 512 * 1024);
  _Float16* wqq  = take(9ull * 128 * 1024);
  _Float16* wqk  = take(9ull * 128 * 1024);
  _Float16* wcq  = take(512ull * 1024);
  _Float16* sk16 = take((size_t)BB * SFN * HW * KDIM);  // keys  [B][SF*hw][KD]
  _Float16* sv16 = take((size_t)BB * VDIM * SFN * HW);  // vals  [B][VD][SF*hw]
  _Float16* qqo  = take((size_t)BB * FF * HW * KDIM);   // qq    [B][F*hw][KD]
  _Float16* qko  = take((size_t)BB * HW * KDIM);        // qk mid[B][hw][KD]
  _Float16* nv16 = take((size_t)BB * VDIM * HW);        // new_V [B][VD][hw]

  // ---- conversions ----
  {
    long t = 10L * 1024 * HW;
    k_nchw2nhwc<<<dim3((unsigned)((t + 255) / 256)), dim3(256), 0, stream>>>(query, q16, 10, 1024);
    t = 30L * 1024 * HW;
    k_nchw2nhwc<<<dim3((unsigned)((t + 255) / 256)), dim3(256), 0, stream>>>(support, s16, 30, 1024);
    t = 128L * 1024 * 9;
    k_wcvt<<<dim3((unsigned)((t + 255) / 256)), dim3(256), 0, stream>>>(w_sk, wsk, 128, 1024, 9);
    k_wcvt<<<dim3((unsigned)((t + 255) / 256)), dim3(256), 0, stream>>>(w_qq, wqq, 128, 1024, 9);
    k_wcvt<<<dim3((unsigned)((t + 255) / 256)), dim3(256), 0, stream>>>(w_qk, wqk, 128, 1024, 9);
    t = 512L * 1024 * 9;
    k_wcvt<<<dim3((unsigned)((t + 255) / 256)), dim3(256), 0, stream>>>(w_sv, wsv, 512, 1024, 9);
    t = 512L * 1024;
    k_wcvt<<<dim3((unsigned)((t + 255) / 256)), dim3(256), 0, stream>>>(w_cq, wcq, 512, 1024, 1);
  }

  const dim3 blk(32);
  // ---- QKV conv projections (implicit-GEMM WMMA, 32px x 64co per wave) ----
  // grid.x = 32 rows * (Cout/64) groups
  // sk: [B][SF*hw][KD]   off = i*hw*KD + p*KD + co
  k_conv_wmma<<<dim3(32 * 2, 30), blk, 0, stream>>>(
      s16, wsk, b_sk, sk16, nullptr, 1024, 128, 9,
      0, 1, 1, (long)HW * KDIM, 0, 1, KDIM);
  // sv: [B][VD][SF*hw]   off = b*VD*SF*hw + sf*hw + co*SF*hw + p
  k_conv_wmma<<<dim3(32 * 8, 30), blk, 0, stream>>>(
      s16, wsv, b_sv, sv16, nullptr, 1024, 512, 9,
      0, 1, SFN, (long)VDIM * SFN * HW, HW, (long)SFN * HW, 1);
  // qq: [B][F*hw][KD]
  k_conv_wmma<<<dim3(32 * 2, 10), blk, 0, stream>>>(
      q16, wqq, b_qq, qqo, nullptr, 1024, 128, 9,
      0, 1, 1, (long)HW * KDIM, 0, 1, KDIM);
  // qk: middle frames only (n = b*F + 2)
  k_conv_wmma<<<dim3(32 * 2, 2), blk, 0, stream>>>(
      q16, wqk, b_qk, qko, nullptr, 1024, 128, 9,
      2, FF, 1, (long)HW * KDIM, 0, 1, KDIM);
  // qf: 1x1 conv, f32 straight into d_out channels [VD..2VD)
  k_conv_wmma<<<dim3(32 * 8, 10), blk, 0, stream>>>(
      q16, wcq, b_cq, nullptr, out + (long)VDIM * HW, 1024, 512, 1,
      0, 1, 1, 2L * VDIM * HW, 0, HW, 1);

  // ---- stage 1: middle-frame queries vs support memory -> new_V ----
  k_flash_wmma<<<dim3(64, 4, BB), blk, 0, stream>>>(
      qqo + 2L * HW * KDIM, (long)FF * HW * KDIM,   // Q = qq middle frame
      sk16, (long)SFN * HW * KDIM,                  // K
      sv16, (long)VDIM * SFN * HW, (long)SFN * HW,  // V (v-major)
      SFN * HW, nv16, nullptr);

  // ---- stage 2: all-frame queries vs middle-frame keys -> d_out[0..VD) ----
  k_flash_wmma<<<dim3(320, 4, BB), blk, 0, stream>>>(
      qqo, (long)FF * HW * KDIM,                    // Q = all qq
      qko, (long)HW * KDIM,                         // K = qk middle
      nv16, (long)VDIM * HW, (long)HW,              // V = new_V
      HW, nullptr, out);
}
